// AdaptivePhysicsMask_6098853560434
// MI455X (gfx1250) — matrix operand order repaired and checked by reading, not compiled, not run on hardware
//
#include <hip/hip_runtime.h>
#include <hip/hip_bf16.h>
#include <math.h>

// ---------------------------------------------------------------------------
// AdaptivePhysicsMask for MI455X (gfx1250, wave32, WMMA)
//
// Dominant cost: [B*N*N, 64] x [64, 64] GEMM (17.2 GFLOP) fully fused so the
// 536 MB "h" tensor never touches HBM (output is only 8 MB -> compute bound).
// GEMM runs on v_wmma_f32_16x16x32_f16 (f16 in, f32 accumulate).
// ---------------------------------------------------------------------------

typedef _Float16 v16h __attribute__((ext_vector_type(16)));
typedef float    v8f  __attribute__((ext_vector_type(8)));
typedef float    v4f  __attribute__((ext_vector_type(4)));

#define GH   32
#define GW   32
#define NP   1024          // GH*GW
#define HID  64
#define BATCH 2
#define HW   128           // image H == W
#define PB   4             // pool block (128/32)

// ---------------------------------------------------------------------------
// Prep 1: pooling + feature build + per-patch linear projections.
//   pi[t][h]  = feat(t) . W1[0:3][h]
//   pjb[t][h] = feat(t) . W1[3:6][h] + b1[h]      (b1 folded in)
//   windp[t]  = avg |wind| over 4x4 block
// t = b*1024 + n, 2048 threads total.
// ---------------------------------------------------------------------------
__global__ __launch_bounds__(256)
void apm_prep1(const float* __restrict__ elev,
               const float* __restrict__ u, const float* __restrict__ v,
               const float* __restrict__ W1, const float* __restrict__ b1,
               float* __restrict__ pi, float* __restrict__ pjb,
               float* __restrict__ windp)
{
    int t = blockIdx.x * blockDim.x + threadIdx.x;
    if (t >= BATCH * NP) return;
    int b = t >> 10;
    int n = t & (NP - 1);
    int gi = n >> 5, gj = n & 31;

    const float* ub = u + b * HW * HW + (gi * PB) * HW + gj * PB;
    const float* vb = v + b * HW * HW + (gi * PB) * HW + gj * PB;

    float su = 0.f, sv = 0.f, sm = 0.f;
#pragma unroll
    for (int r = 0; r < PB; ++r) {
#pragma unroll
        for (int c = 0; c < PB; ++c) {
            float uu = ub[r * HW + c];
            float vv = vb[r * HW + c];
            su += uu; sv += vv;
            sm += sqrtf(uu * uu + vv * vv);
        }
    }
    const float inv16 = 1.0f / 16.0f;
    float up = su * inv16, vp = sv * inv16, wp = sm * inv16;
    windp[t] = wp;
    float f2 = elev[t] * 1e-3f;

    float* pio = pi  + t * HID;
    float* pjo = pjb + t * HID;
#pragma unroll 8
    for (int h = 0; h < HID; ++h) {
        pio[h] = up * W1[h] + vp * W1[HID + h] + f2 * W1[2 * HID + h];
        pjo[h] = up * W1[3 * HID + h] + vp * W1[4 * HID + h]
               + f2 * W1[5 * HID + h] + b1[h];
    }
}

// ---------------------------------------------------------------------------
// Prep 2: W2 [k][n] f32  ->  W2T [n][k] f16  (B-operand friendly: K contiguous)
// ---------------------------------------------------------------------------
__global__ __launch_bounds__(256)
void apm_prep2(const float* __restrict__ W2, _Float16* __restrict__ W2T)
{
    int t = blockIdx.x * blockDim.x + threadIdx.x;
    if (t >= HID * HID) return;
    int n = t >> 6, k = t & 63;
    W2T[t] = (_Float16)W2[k * HID + n];
}

// ---------------------------------------------------------------------------
// Main: one wave per 16x16 (i,j) output tile.  8 waves / 256-thread block.
// Per M-tile (fixed i, 16 j's):  A = f16(relu(pi_i + pjb_j)), 8 WMMAs,
// register epilogue (relu(+b2).W3, shfl-reduce, tanh, physics bias, clip).
// ---------------------------------------------------------------------------
__global__ __launch_bounds__(256)
void apm_main(const float* __restrict__ elev,
              const float* __restrict__ b2, const float* __restrict__ W3,
              const float* __restrict__ b3, const float* __restrict__ alpha,
              const float* __restrict__ pi, const float* __restrict__ pjb,
              const float* __restrict__ windp,
              const _Float16* __restrict__ W2T,
              float* __restrict__ out)
{
    const int lane = threadIdx.x & 31;
    const int wid  = blockIdx.x * (blockDim.x >> 5) + (threadIdx.x >> 5);
    const int b    = wid >> 12;            // / (64*64)
    const int it   = (wid >> 6) & 63;
    const int jt   = wid & 63;
    const int i0   = it << 4;
    const int j0   = jt << 4;
    const int l16  = lane & 15;
    const int hi   = lane >> 4;            // 0: lanes 0-15, 1: lanes 16-31

    // ---- preload B operands of W2 (reused by all 16 M-tiles) --------------
    // 16-bit B 32x16 layout: lane = column (lanes 0-15 -> K 0..7,16..23 of the
    // k-step; lanes 16-31 -> K 8..15,24..31), K contiguous per lane in W2T.
    v16h Bop[4][2];
#pragma unroll
    for (int nt = 0; nt < 4; ++nt) {
        int n = nt * 16 + l16;
#pragma unroll
        for (int ks = 0; ks < 2; ++ks) {
            int kb = ks * 32 + hi * 8;
            union { v4f f[2]; v16h h; } uu;
            const v4f* p = (const v4f*)(W2T + n * HID + kb);
            uu.f[0] = p[0];
            uu.f[1] = p[2];
            Bop[nt][ks] = uu.h;
        }
    }

    // ---- this lane's K-subset of pjb for its j row -------------------------
    // A-operand lane needs K blocks (2q+hi)*8, q=0..3 -> 32 of 64 values.
    float pjv[32];
    {
        const float* base = pjb + (((b << 10) + j0 + l16) << 6) + (hi << 3);
#pragma unroll
        for (int q = 0; q < 4; ++q) {
            v4f a = *(const v4f*)(base + q * 16);
            v4f c = *(const v4f*)(base + q * 16 + 4);
#pragma unroll
            for (int e = 0; e < 4; ++e) {
                pjv[q * 8 + e]     = a[e];
                pjv[q * 8 + 4 + e] = c[e];
            }
        }
    }

    // ---- per-lane epilogue constants --------------------------------------
    float b2v[4], w3v[4];
#pragma unroll
    for (int nt = 0; nt < 4; ++nt) {
        b2v[nt] = b2[nt * 16 + l16];
        w3v[nt] = W3[nt * 16 + l16];
    }
    const float b3s = b3[0];
    const float al  = alpha[0];
    const float* elevb = elev  + (b << 10);
    const float* wpb   = windp + (b << 10);

    // ---- 16 M-tiles: fixed i, rows = 16 j's -------------------------------
    for (int t = 0; t < 16; ++t) {
        const int i = i0 + t;

        // h1 = relu(pi_i + pjb_j) on this lane's K subset
        float h1[32];
        const float* pib = pi + (((b << 10) + i) << 6) + (hi << 3);
#pragma unroll
        for (int q = 0; q < 4; ++q) {
            v4f a = *(const v4f*)(pib + q * 16);
            v4f c = *(const v4f*)(pib + q * 16 + 4);
#pragma unroll
            for (int e = 0; e < 4; ++e) {
                h1[q * 8 + e]     = fmaxf(a[e] + pjv[q * 8 + e],     0.f);
                h1[q * 8 + 4 + e] = fmaxf(c[e] + pjv[q * 8 + 4 + e], 0.f);
            }
        }

        // Pack into WMMA A-registers.  With the block-interleaved subset
        // order above, element e of k-step ks is simply h1[e + 16*ks].
        v16h A0, A1;
#pragma unroll
        for (int e = 0; e < 16; ++e) {
            A0[e] = (_Float16)h1[e];
            A1[e] = (_Float16)h1[16 + e];
        }

        v8f acc[4];
#pragma unroll
        for (int nt = 0; nt < 4; ++nt)
#pragma unroll
            for (int e = 0; e < 8; ++e) acc[nt][e] = 0.f;

#pragma unroll
        for (int nt = 0; nt < 4; ++nt) {
            acc[nt] = __builtin_amdgcn_wmma_f32_16x16x32_f16(
                false, A0, false, Bop[nt][0], (short)0, acc[nt], false, false);
            acc[nt] = __builtin_amdgcn_wmma_f32_16x16x32_f16(
                false, A1, false, Bop[nt][1], (short)0, acc[nt], false, false);
        }

        // Epilogue: relu(h2 + b2) . W3, reduce across the 16 columns held by
        // the 16 lanes of each half-wave.
        float s[8];
#pragma unroll
        for (int vv = 0; vv < 8; ++vv) {
            float a = 0.f;
#pragma unroll
            for (int nt = 0; nt < 4; ++nt)
                a += fmaxf(acc[nt][vv] + b2v[nt], 0.f) * w3v[nt];
            s[vv] = a;
        }
#pragma unroll
        for (int m = 1; m <= 8; m <<= 1) {
#pragma unroll
            for (int vv = 0; vv < 8; ++vv)
                s[vv] += __shfl_xor(s[vv], m, 32);
        }

        // Physics bias + store.  Rows 0-7 live in the low half-wave, 8-15 in
        // the high half; lane 0 / lane 16 write 8 outputs each.
        const float ei = elevb[i];
        const float wi = wpb[i];
        if (l16 == 0) {
            const long orow = ((long)((b << 10) + i) << 10);
#pragma unroll
            for (int vv = 0; vv < 8; ++vv) {
                int j = j0 + (hi << 3) + vv;
                float ej = elevb[j];
                float wj = wpb[j];
                float corr = al * tanhf(s[vv] + b3s);
                float ed   = (ej - ei) * 1e-3f;
                float eb   = -0.5f * fmaxf(ed, 0.f);
                float wmin = fminf(wi, wj);
                float sig  = 1.0f / (1.0f + __expf(5.0f - wmin));
                float fb   = eb * (1.0f - 0.3f * sig);
                fb = fminf(fmaxf(fb, -10.0f), 0.0f);
                float ov = fminf(fmaxf(fb + corr, -10.0f), 10.0f);
                out[orow + j] = ov;
            }
        }
    }
}

// ---------------------------------------------------------------------------
// Launch.  ws layout (floats): pi[131072] | pjb[131072] | windp[2048] |
// W2T (f16, 4096 halves).  Total ~1.06 MB.
// ---------------------------------------------------------------------------
extern "C" void kernel_launch(void* const* d_in, const int* in_sizes, int n_in,
                              void* d_out, int out_size, void* d_ws, size_t ws_size,
                              hipStream_t stream)
{
    const float* elev  = (const float*)d_in[0];
    const float* u     = (const float*)d_in[1];
    const float* v     = (const float*)d_in[2];
    const float* W1    = (const float*)d_in[3];
    const float* b1    = (const float*)d_in[4];
    const float* W2    = (const float*)d_in[5];
    const float* b2    = (const float*)d_in[6];
    const float* W3    = (const float*)d_in[7];
    const float* b3    = (const float*)d_in[8];
    const float* alpha = (const float*)d_in[9];

    float* ws      = (float*)d_ws;
    float* pi      = ws;                    // 131072 f32
    float* pjb     = ws + 131072;           // 131072 f32
    float* windp   = ws + 262144;           // 2048 f32
    _Float16* W2T  = (_Float16*)(ws + 264192); // 4096 f16

    apm_prep1<<<8, 256, 0, stream>>>(elev, u, v, W1, b1, pi, pjb, windp);
    apm_prep2<<<16, 256, 0, stream>>>(W2, W2T);

    // 2 * 64 * 64 = 8192 waves, 8 waves per 256-thread block.
    apm_main<<<1024, 256, 0, stream>>>(elev, b2, W3, b3, alpha,
                                       pi, pjb, windp, W2T, (float*)d_out);
}
